// RegionProposalNetwork_34273839022583
// MI455X (gfx1250) — matrix operand order repaired and checked by reading, not compiled, not run on hardware
//
#include <hip/hip_runtime.h>

typedef __bf16 bf16_t;
typedef __attribute__((ext_vector_type(16))) __bf16 v16bf;
typedef __attribute__((ext_vector_type(8)))  __bf16 v8bf;
typedef __attribute__((ext_vector_type(8)))  float  v8f;
typedef __attribute__((ext_vector_type(4)))  unsigned int v4u;
typedef __attribute__((ext_vector_type(8)))  int v8i_t;
typedef __attribute__((ext_vector_type(4)))  int v4i_t;

#define DEVINL __device__ __forceinline__

#if defined(__has_builtin)
#if __has_builtin(__builtin_amdgcn_tensor_load_to_lds)
#define HAVE_TDM 1
#endif
#endif

// ---------------------------------------------------------------------------
// conv1: direct f32 conv 7x7 stride4 pad3, 3->64 on 1600x1600, ReLU -> bf16
// ---------------------------------------------------------------------------
__global__ void conv1_kernel(const float* __restrict__ img, const float* __restrict__ w1,
                             const float* __restrict__ b1, bf16_t* __restrict__ out)
{
    int idx = blockIdx.x * 256 + threadIdx.x;
    if (idx >= 64 * 400 * 400) return;
    int co  = idx / 160000;
    int rem = idx - co * 160000;
    int oy  = rem / 400, ox = rem - oy * 400;
    float sum = b1[co];
    const float* wbase = w1 + co * 147;
    for (int ci = 0; ci < 3; ++ci) {
        const float* ibase = img + (size_t)ci * 2560000;
        for (int ky = 0; ky < 7; ++ky) {
            int iy = oy * 4 - 3 + ky;
            if (iy < 0 || iy >= 1600) continue;
            for (int kx = 0; kx < 7; ++kx) {
                int ix = ox * 4 - 3 + kx;
                if (ix < 0 || ix >= 1600) continue;
                sum += ibase[iy * 1600 + ix] * wbase[ci * 49 + ky * 7 + kx];
            }
        }
    }
    out[idx] = (bf16_t)fmaxf(sum, 0.f);
}

// ---------------------------------------------------------------------------
// weight reorder + f32->bf16: out[co][(ky*3+kx)*CIN + ci] = w[co][ci][ky][kx]
// (makes each 32-wide K chunk have constant (ky,kx) and contiguous ci)
// ---------------------------------------------------------------------------
template<int CIN>
__global__ void wreorder_kernel(const float* __restrict__ w, bf16_t* __restrict__ out)
{
    constexpr int KTOT = CIN * 9;
    int idx = blockIdx.x * 256 + threadIdx.x;
    if (idx >= 256 * KTOT) return;
    int co = idx / KTOT;
    int r  = idx - co * KTOT;
    int q  = r / CIN;            // CIN is pow2 -> shift
    int ci = r & (CIN - 1);
    out[idx] = (bf16_t)w[(size_t)co * KTOT + ci * 9 + q];
}

// ---------------------------------------------------------------------------
// Tensor Data Mover: 2D tile (tile_x elems of 2B per row, tile_y rows) -> LDS
// Descriptor packing per CDNA5 ISA.
// ---------------------------------------------------------------------------
DEVINL void tdm_issue_2d(unsigned lds_off, const bf16_t* gptr,
                         int tile_x, int tile_y, int tensor_x, int tensor_y,
                         int stride_elems)
{
#ifdef HAVE_TDM
    unsigned long long ga = (unsigned long long)(uintptr_t)gptr;
    v4u g0;
    g0.x = 1u;                                        // count=1, user mode
    g0.y = lds_off;                                   // lds_addr
    g0.z = (unsigned)(ga & 0xFFFFFFFFu);              // global_addr[31:0]
    g0.w = (unsigned)((ga >> 32) & 0x01FFFFFFu) | (2u << 30);  // addr[56:32] | type=2
    v8i_t g1;
    g1[0] = (1 << 16);                                // wg_mask=0, data_size=1 (2B)
    g1[1] = (int)(((unsigned)tensor_x & 0xFFFFu) << 16);            // tensor_dim0 lo16
    g1[2] = (int)((((unsigned)tensor_x >> 16) & 0xFFFFu)
                  | (((unsigned)tensor_y & 0xFFFFu) << 16));        // dim0 hi | dim1 lo
    g1[3] = (int)((((unsigned)tensor_y >> 16) & 0xFFFFu)
                  | (((unsigned)tile_x & 0xFFFFu) << 16));          // dim1 hi | tile_dim0
    g1[4] = (int)((unsigned)tile_y & 0xFFFFu);                      // tile_dim1 | tile_dim2=0
    g1[5] = stride_elems;                             // tensor_dim0_stride[31:0]
    g1[6] = 0;                                        // stride0 hi | stride1 lo
    g1[7] = 0;
    v4i_t z4 = {0, 0, 0, 0};
#if __clang_major__ >= 23
    v8i_t z8 = {0, 0, 0, 0, 0, 0, 0, 0};
    __builtin_amdgcn_tensor_load_to_lds(g0, g1, z4, z4, z8, 0);
#else
    __builtin_amdgcn_tensor_load_to_lds(g0, g1, z4, z4, 0);
#endif
#endif
}

// ---------------------------------------------------------------------------
// 3x3 conv as implicit GEMM on V_WMMA_F32_16X16X32_BF16.
// Block: 1024 thr (32 waves) -> 8x8 pixel M-tile x 128-channel N-tile.
// K order: (ky*3+kx)*CIN + ci. A scattered into fragment layout in LDS.
// B panel (128 x 32 bf16) double-buffered, fetched by TDM one step ahead.
// ---------------------------------------------------------------------------
template<int CIN, int SS, int PP, int WIN, int HIN>
__global__ void conv3x3_wmma(const bf16_t* __restrict__ in, const bf16_t* __restrict__ wt,
                             const float* __restrict__ bias, float* __restrict__ outF,
                             bf16_t* __restrict__ outBF)
{
    constexpr int KTOT  = CIN * 9;
    constexpr int STEPS = KTOT / 32;
    constexpr int CILOG = (CIN == 64) ? 6 : 8;

    __shared__ alignas(64) bf16_t ldsAf[4][32][16];     // [msub][lane][elem] 4KB
    __shared__ alignas(64) bf16_t ldsB[2][128][32];     // double-buffered B   16KB

    const int ox0  = blockIdx.x * 8;
    const int oy0  = blockIdx.y * 8;
    const int bz   = blockIdx.z;            // 128-channel group
    const int tid  = threadIdx.x;
    const int lane = tid & 31;
    const int wv   = tid >> 5;
    const int half = lane >> 4;
    const int nn   = lane & 15;
    const int msub = wv >> 3;
    const int ni   = wv & 7;

    const bool edge = (blockIdx.x == 0) | (blockIdx.x >= 12) |
                      (blockIdx.y == 0) | (blockIdx.y >= 12);

    const bf16_t* wpanel = wt + (size_t)(bz * 128) * KTOT;

#ifdef HAVE_TDM
    constexpr bool USE_TDM = true;
#else
    constexpr bool USE_TDM = false;
#endif

    // ---- B stage: TDM issue (wave0) or cooperative vector copy (all) ----
    auto stage_b = [&](int buf, int kb) {
        if (USE_TDM) {
            unsigned lds_off = (unsigned)(uintptr_t)&ldsB[buf][0][0];
            tdm_issue_2d(lds_off, wpanel + kb, 32, 128, KTOT, 128, KTOT);
        } else {
            if (tid < 512) {
                int co = tid >> 2, ko = (tid & 3) * 8;
                *(v8bf*)&ldsB[buf][co][ko] =
                    *(const v8bf*)(wpanel + (size_t)co * KTOT + kb + ko);
            }
        }
    };

    v8f acc = {0.f, 0.f, 0.f, 0.f, 0.f, 0.f, 0.f, 0.f};

    int buf = 0;
    if (!USE_TDM || wv == 0) stage_b(0, 0);

    for (int s = 0; s < STEPS; ++s) {
        const int kb  = s * 32;
        const int q   = kb >> CILOG;         // (ky*3+kx), constant over chunk
        const int ci0 = kb & (CIN - 1);
        const int ky  = q / 3, kx = q - ky * 3;

        // ---- stage A chunk directly into fragment layout ----
        #pragma unroll
        for (int rep = 0; rep < 2; ++rep) {
            int e   = tid + rep * 1024;      // 0..2047
            int t   = e >> 6;                // K offset 0..31
            int pix = e & 63;
            int dy  = pix >> 3, dx = pix & 7;
            int iy  = (oy0 + dy) * SS - PP + ky;
            int ix  = (ox0 + dx) * SS - PP + kx;
            int ci  = ci0 + t;
            bf16_t v;
            if (!edge) {
                v = in[(size_t)ci * (WIN * HIN) + iy * WIN + ix];
            } else {
                v = (bf16_t)0.f;
                if (ix >= 0 && ix < WIN && iy >= 0 && iy < HIN)
                    v = in[(size_t)ci * (WIN * HIN) + iy * WIN + ix];
            }
            int h  = t >> 4;
            int i  = t & 15;
            ldsAf[pix >> 4][(pix & 15) + 16 * h][i] = v;
        }

#ifdef HAVE_TDM
        if (wv == 0) __builtin_amdgcn_s_wait_tensorcnt((short)0);
#endif
        __syncthreads();

        // prefetch next B panel into other buffer
        if (s + 1 < STEPS) {
            if (!USE_TDM || wv == 0) stage_b(buf ^ 1, kb + 32);
        }

        // ---- fragments + WMMA ----
        v16bf a = *(const v16bf*)&ldsAf[msub][lane][0];
        v16bf b = *(const v16bf*)&ldsB[buf][ni * 16 + nn][half * 16];
        acc = __builtin_amdgcn_wmma_f32_16x16x32_bf16(false, a, false, b,
                                                      (short)0, acc, false, false);
        __syncthreads();
        buf ^= 1;
    }

    // ---- writeback: D layout VGPR r -> M = r + 8*half, N = lane&15 ----
    const int co = bz * 128 + ni * 16 + nn;
    const float bv = bias[co];
    #pragma unroll
    for (int r = 0; r < 8; ++r) {
        int p  = msub * 16 + r + half * 8;   // pixel index 0..63
        int oy = oy0 + (p >> 3);
        int ox = ox0 + (p & 7);
        if (ox < 100 && oy < 100) {
            float v = fmaxf(acc[r] + bv, 0.f);
            size_t o = (size_t)co * 10000 + oy * 100 + ox;
            outF[o] = v;
            if (outBF) outBF[o] = (bf16_t)v;
        }
    }
}

// ---------------------------------------------------------------------------
// 1x1 heads: one thread per pixel, 18+36 accumulators, weights scalarized
// ---------------------------------------------------------------------------
__global__ void heads_kernel(const float* __restrict__ h,
                             const float* __restrict__ w_cls, const float* __restrict__ b_cls,
                             const float* __restrict__ w_reg, const float* __restrict__ b_reg,
                             float* __restrict__ cls_out, float* __restrict__ reg_out)
{
    int p = blockIdx.x * 256 + threadIdx.x;
    if (p >= 10000) return;
    float ac[18], ar[36];
    #pragma unroll
    for (int c = 0; c < 18; ++c) ac[c] = b_cls[c];
    #pragma unroll
    for (int c = 0; c < 36; ++c) ar[c] = b_reg[c];
    for (int ci = 0; ci < 256; ++ci) {
        float hv = h[(size_t)ci * 10000 + p];
        #pragma unroll
        for (int c = 0; c < 18; ++c) ac[c] += hv * w_cls[c * 256 + ci];
        #pragma unroll
        for (int c = 0; c < 36; ++c) ar[c] += hv * w_reg[c * 256 + ci];
    }
    #pragma unroll
    for (int c = 0; c < 18; ++c) cls_out[(size_t)c * 10000 + p] = ac[c];
    #pragma unroll
    for (int c = 0; c < 36; ++c) reg_out[(size_t)c * 10000 + p] = ar[c];
}

// ---------------------------------------------------------------------------
// anchors / IoU
// ---------------------------------------------------------------------------
DEVINL void anchor_box(int n, float& x1, float& y1, float& x2, float& y2)
{
    const float sizes[3]  = {128.f, 256.f, 512.f};
    const float ratios[3] = {0.5f, 1.f, 2.f};
    int a = n % 9;  int t = n / 9;
    int y = t % 100, x = t / 100;
    float sq = sqrtf(ratios[a % 3]);
    float w = sizes[a / 3] * sq, h = sizes[a / 3] / sq;
    float cx = (x + 0.5f) * 16.f, cy = (y + 0.5f) * 16.f;
    x1 = cx - 0.5f * w; y1 = cy - 0.5f * h; x2 = x1 + w; y2 = y1 + h;
}

DEVINL float iou1(float ax1, float ay1, float ax2, float ay2, const float* g)
{
    float lx = fmaxf(ax1, g[0]), ly = fmaxf(ay1, g[1]);
    float rx = fminf(ax2, g[2]), ry = fminf(ay2, g[3]);
    float inter = fmaxf(rx - lx, 0.f) * fmaxf(ry - ly, 0.f);
    float aa = (ax2 - ax1) * (ay2 - ay1);
    float ab = (g[2] - g[0]) * (g[3] - g[1]);
    return inter / (aa + ab - inter);
}

__global__ void zero_gtmax_kernel(unsigned* gt_max)
{
    if (threadIdx.x < 64) gt_max[threadIdx.x] = 0u;
}

__global__ void anchor_pass1(const float* __restrict__ gt, float* __restrict__ anc_max,
                             int* __restrict__ anc_arg, unsigned* __restrict__ gt_max)
{
    __shared__ unsigned lmax[64];
    if (threadIdx.x < 64) lmax[threadIdx.x] = 0u;
    __syncthreads();
    int n = blockIdx.x * 256 + threadIdx.x;
    if (n < 90000) {
        float x1, y1, x2, y2; anchor_box(n, x1, y1, x2, y2);
        float best = -1.f; int bi = 0;
        for (int m = 0; m < 64; ++m) {
            float v = iou1(x1, y1, x2, y2, gt + 4 * m);
            if (v > best) { best = v; bi = m; }
            atomicMax(&lmax[m], __float_as_uint(v));   // v>=0: uint order == float order
        }
        anc_max[n] = best; anc_arg[n] = bi;
    }
    __syncthreads();
    if (threadIdx.x < 64) atomicMax(&gt_max[threadIdx.x], lmax[threadIdx.x]);
}

__global__ void anchor_pass2(const float* __restrict__ gt, const float* __restrict__ anc_max,
                             const unsigned* __restrict__ gt_max,
                             float* __restrict__ key_pos, float* __restrict__ key_neg)
{
    int n = blockIdx.x * 256 + threadIdx.x;
    if (n >= 90000) return;
    float x1, y1, x2, y2; anchor_box(n, x1, y1, x2, y2);
    bool best = false;
    for (int m = 0; m < 64; ++m) {
        float v = iou1(x1, y1, x2, y2, gt + 4 * m);
        if (__float_as_uint(v) == gt_max[m]) best = true;
    }
    float am = anc_max[n];
    float label = -1.f;
    if (am < 0.2f) label = 0.f;
    if (best)      label = 1.f;
    if (am > 0.7f) label = 1.f;
    key_pos[n] = (label == 1.f) ? am : -1.f;
    key_neg[n] = (label == 0.f) ? (1.0e6f - (float)n) : -1.0e9f;
}

// ---------------------------------------------------------------------------
// exact stable top-32 (matches jax.lax.top_k: lower index wins ties)
// ---------------------------------------------------------------------------
__global__ void topk_kernel(const float* __restrict__ key, int* __restrict__ out)
{
    __shared__ float sv[1024];
    __shared__ int   si[1024];
    __shared__ float pV;
    __shared__ int   pI;
    int tid = threadIdx.x;
    if (tid == 0) { pV = 3.0e38f; pI = -1; }
    __syncthreads();
    for (int j = 0; j < 32; ++j) {
        float prevV = pV; int prevI = pI;
        float bv = -3.0e38f; int bi = 1 << 30;
        for (int n = tid; n < 90000; n += 1024) {
            float v = key[n];
            bool lt = (v < prevV) || (v == prevV && n > prevI);
            if (lt && (v > bv || (v == bv && n < bi))) { bv = v; bi = n; }
        }
        sv[tid] = bv; si[tid] = bi;
        __syncthreads();
        for (int s = 512; s > 0; s >>= 1) {
            if (tid < s) {
                float ov = sv[tid + s]; int oi = si[tid + s];
                if (ov > sv[tid] || (ov == sv[tid] && oi < si[tid])) { sv[tid] = ov; si[tid] = oi; }
            }
            __syncthreads();
        }
        if (tid == 0) { out[j] = si[0]; pV = sv[0]; pI = si[0]; }
        __syncthreads();
    }
}

// ---------------------------------------------------------------------------
// final gather + losses
// out layout: [0]=loss, [1..2560000]=fm, proposals(128), pos_index(32), gt_cls(32)
// ---------------------------------------------------------------------------
#define OUT_PROP 2560001
#define OUT_POS  2560129
#define OUT_GT   2560161

__global__ void finalize_kernel(const float* __restrict__ gtb, const int* __restrict__ gtc,
                                const int* __restrict__ anc_arg,
                                const int* __restrict__ pos_idx, const int* __restrict__ neg_idx,
                                const float* __restrict__ cls_out, const float* __restrict__ reg_out,
                                float* __restrict__ out)
{
    if (threadIdx.x != 0 || blockIdx.x != 0) return;
    float logits[64][2];
    float reg_sum = 0.f;
    for (int j = 0; j < 32; ++j) {
        int n = pos_idx[j];
        int a = n % 9; int t = n / 9; int y = t % 100; int x = t / 100;
        float x1, y1, x2, y2; anchor_box(n, x1, y1, x2, y2);
        logits[j][0] = cls_out[(a * 2 + 0) * 10000 + y * 100 + x];
        logits[j][1] = cls_out[(a * 2 + 1) * 10000 + y * 100 + x];
        float off[4];
        for (int c = 0; c < 4; ++c) off[c] = reg_out[(a * 4 + c) * 10000 + y * 100 + x];
        float w = x2 - x1, h = y2 - y1;
        float bcx = x1 + 0.5f * w, bcy = y1 + 0.5f * h;
        float ncx = bcx + off[0] * w, ncy = bcy + off[1] * h;
        float nw = w * expf(off[2]), nh = h * expf(off[3]);
        out[OUT_PROP + j * 4 + 0] = ncx - 0.5f * nw;
        out[OUT_PROP + j * 4 + 1] = ncy - 0.5f * nh;
        out[OUT_PROP + j * 4 + 2] = ncx + 0.5f * nw;
        out[OUT_PROP + j * 4 + 3] = ncy + 0.5f * nh;
        int gm = anc_arg[n];
        float gx1 = gtb[gm * 4 + 0], gy1 = gtb[gm * 4 + 1];
        float gx2 = gtb[gm * 4 + 2], gy2 = gtb[gm * 4 + 3];
        float gw = gx2 - gx1, gh = gy2 - gy1;
        float gcx = gx1 + 0.5f * gw, gcy = gy1 + 0.5f * gh;
        float tt[4] = {(gcx - bcx) / w, (gcy - bcy) / h, logf(gw / w), logf(gh / h)};
        for (int c = 0; c < 4; ++c) {
            float d = tt[c] - off[c], ad = fabsf(d);
            reg_sum += (ad < 1.f) ? 0.5f * d * d : ad - 0.5f;
        }
        out[OUT_POS + j] = (float)n;
        out[OUT_GT + j]  = (float)gtc[gm];
    }
    for (int j = 0; j < 32; ++j) {
        int n = neg_idx[j];
        int a = n % 9; int t = n / 9; int y = t % 100; int x = t / 100;
        logits[32 + j][0] = cls_out[(a * 2 + 0) * 10000 + y * 100 + x];
        logits[32 + j][1] = cls_out[(a * 2 + 1) * 10000 + y * 100 + x];
    }
    float cls_sum = 0.f;
    for (int r = 0; r < 64; ++r) {
        int tgt = (r < 32) ? 1 : 0;
        float l0 = logits[r][0], l1 = logits[r][1];
        float m = fmaxf(l0, l1);
        float lse = m + logf(expf(l0 - m) + expf(l1 - m));
        cls_sum += lse - logits[r][tgt];
    }
    out[0] = (cls_sum / 64.f) + 5.f * (reg_sum / 128.f);
}

// ---------------------------------------------------------------------------
extern "C" void kernel_launch(void* const* d_in, const int* in_sizes, int n_in,
                              void* d_out, int out_size, void* d_ws, size_t ws_size,
                              hipStream_t stream)
{
    const float* img   = (const float*)d_in[0];
    const float* gtb   = (const float*)d_in[1];
    const int*   gtc   = (const int*)  d_in[2];
    const float* w1    = (const float*)d_in[3];
    const float* b1    = (const float*)d_in[4];
    const float* w2    = (const float*)d_in[5];
    const float* b2    = (const float*)d_in[6];
    const float* w_rpn = (const float*)d_in[7];
    const float* b_rpn = (const float*)d_in[8];
    const float* w_cls = (const float*)d_in[9];
    const float* b_cls = (const float*)d_in[10];
    const float* w_reg = (const float*)d_in[11];
    const float* b_reg = (const float*)d_in[12];
    float* out = (float*)d_out;

    char* ws = (char*)d_ws;
    size_t off = 0;
    auto carve = [&](size_t bytes) -> void* {
        void* p = ws + off;
        off = (off + bytes + 255) & ~(size_t)255;
        return p;
    };
    bf16_t*   fm1_bf  = (bf16_t*)  carve((size_t)64 * 400 * 400 * 2);
    bf16_t*   w2_bf   = (bf16_t*)  carve((size_t)256 * 576 * 2);
    bf16_t*   wrpn_bf = (bf16_t*)  carve((size_t)256 * 2304 * 2);
    bf16_t*   fm_bf   = (bf16_t*)  carve((size_t)256 * 10000 * 2);
    float*    h       = (float*)   carve((size_t)256 * 10000 * 4);
    float*    cls_out = (float*)   carve((size_t)18 * 10000 * 4);
    float*    reg_out = (float*)   carve((size_t)36 * 10000 * 4);
    float*    anc_max = (float*)   carve(90000 * 4);
    int*      anc_arg = (int*)     carve(90000 * 4);
    float*    key_pos = (float*)   carve(90000 * 4);
    float*    key_neg = (float*)   carve(90000 * 4);
    unsigned* gt_max  = (unsigned*)carve(64 * 4);
    int*      pos_idx = (int*)     carve(32 * 4);
    int*      neg_idx = (int*)     carve(32 * 4);

    // feature extractor
    conv1_kernel<<<(64 * 400 * 400 + 255) / 256, 256, 0, stream>>>(img, w1, b1, fm1_bf);
    wreorder_kernel<64><<<(256 * 576 + 255) / 256, 256, 0, stream>>>(w2, w2_bf);
    wreorder_kernel<256><<<(256 * 2304 + 255) / 256, 256, 0, stream>>>(w_rpn, wrpn_bf);

    // conv2: 64->256 s4 p1 on 400x400 -> fm (float, into d_out+1) + bf16 copy
    conv3x3_wmma<64, 4, 1, 400, 400><<<dim3(13, 13, 2), 1024, 0, stream>>>(
        fm1_bf, w2_bf, b2, out + 1, fm_bf);
    // RPN conv: 256->256 s1 p1 on 100x100 -> h (float)
    conv3x3_wmma<256, 1, 1, 100, 100><<<dim3(13, 13, 2), 1024, 0, stream>>>(
        fm_bf, wrpn_bf, b_rpn, h, (bf16_t*)nullptr);

    // heads (cls 18ch + reg 36ch fused)
    heads_kernel<<<(10000 + 255) / 256, 256, 0, stream>>>(h, w_cls, b_cls, w_reg, b_reg,
                                                          cls_out, reg_out);

    // anchor matching
    zero_gtmax_kernel<<<1, 64, 0, stream>>>(gt_max);
    anchor_pass1<<<(90000 + 255) / 256, 256, 0, stream>>>(gtb, anc_max, anc_arg, gt_max);
    anchor_pass2<<<(90000 + 255) / 256, 256, 0, stream>>>(gtb, anc_max, gt_max, key_pos, key_neg);

    // sampling
    topk_kernel<<<1, 1024, 0, stream>>>(key_pos, pos_idx);
    topk_kernel<<<1, 1024, 0, stream>>>(key_neg, neg_idx);

    // losses + small outputs
    finalize_kernel<<<1, 1, 0, stream>>>(gtb, gtc, anc_arg, pos_idx, neg_idx,
                                         cls_out, reg_out, out);
}
